// MambaEncoder_19748259627344
// MI455X (gfx1250) — matrix operand order repaired and checked by reading, not compiled, not run on hardware
//
#include <hip/hip_runtime.h>
#include <hip/hip_bf16.h>
#include <math.h>

#define D_MODEL 512
#define N_LAYERS 4
#define D_STATE 16
#define D_CONV 4
#define D_INNER 1024
#define DT_RANK 32
#define BATCH 4
#define SEQ 2048
#define M_ROWS (BATCH * SEQ)   // 8192 GEMM rows
#define ASTRIDE 40             // LDS row stride in halves (80B): conflict-free, 16B aligned

typedef __attribute__((ext_vector_type(16))) _Float16 v16h;
typedef __attribute__((ext_vector_type(8)))  _Float16 v8h;
typedef __attribute__((ext_vector_type(8)))  float    v8f;

// ---------------------------------------------------------------------------
// fp32 -> fp16 conversion (weights, every call; deterministic)
// ---------------------------------------------------------------------------
__global__ __launch_bounds__(256)
void f32_to_f16_kernel(const float* __restrict__ in, _Float16* __restrict__ out, int n) {
  int i = blockIdx.x * 256 + threadIdx.x;
  if (i < n) out[i] = (_Float16)in[i];
}

// ---------------------------------------------------------------------------
// LayerNorm over D_MODEL=512, one wave32 per row (16 elems/lane), f16 output
// ---------------------------------------------------------------------------
__global__ __launch_bounds__(256)
void layernorm_f16_kernel(const float* __restrict__ x,
                          const float* __restrict__ w,
                          const float* __restrict__ b,
                          _Float16* __restrict__ out) {
  int row  = blockIdx.x * 8 + (threadIdx.x >> 5);
  int lane = threadIdx.x & 31;
  const float* xr = x + (size_t)row * D_MODEL;
  float v[16];
  float s = 0.f, s2 = 0.f;
#pragma unroll
  for (int i = 0; i < 16; ++i) {
    v[i] = xr[lane + 32 * i];
    s += v[i]; s2 += v[i] * v[i];
  }
#pragma unroll
  for (int off = 16; off >= 1; off >>= 1) {
    s  += __shfl_xor(s,  off, 32);
    s2 += __shfl_xor(s2, off, 32);
  }
  float mu  = s * (1.0f / D_MODEL);
  float var = s2 * (1.0f / D_MODEL) - mu * mu;
  float rs  = rsqrtf(var + 1e-5f);
  _Float16* orow = out + (size_t)row * D_MODEL;
#pragma unroll
  for (int i = 0; i < 16; ++i) {
    int c = lane + 32 * i;
    orow[c] = (_Float16)((v[i] - mu) * rs * w[c] + b[c]);
  }
}

// ---------------------------------------------------------------------------
// WMMA GEMM: out[M,N] = A[M,K](f16,row) * W[N,K](f16,row)^T  (+ epilogue)
//   EP=1: f16 store   EP=2: softplus(acc+bias) f32   EP=3: acc + residual f32
// MW = wave-groups along M (1 or 2); waves along N = 8/MW. Block tile is
// (64*MW) x (16 * 8/MW); every wave is always active -> no exec masking.
// A tile staged in LDS (double-buffered, padded stride 40h); B fragment
// register double-buffered. sched_barrier(0) forces all 8 ds_load_b128 in
// flight before the 4 back-to-back WMMAs (staged dscnt waits, not 0-waits).
// ---------------------------------------------------------------------------
template <int EP, int MW>
__global__ __launch_bounds__(256)
void gemm_wmma_kernel(const _Float16* __restrict__ A,
                      const _Float16* __restrict__ Bw,
                      int K, int lda, int ldb, int ldo,
                      float* __restrict__ outF, _Float16* __restrict__ outH,
                      const float* __restrict__ bias,
                      const float* __restrict__ res) {
  constexpr int NWN = 8 / MW;          // waves along N
  constexpr int MT  = 64 * MW;         // block M tile
  __shared__ __align__(16) _Float16 asmem[2][MT * ASTRIDE];

  const int tid  = threadIdx.x;
  const int wave = tid >> 5;
  const int lane = tid & 31;
  const int hi16 = lane >> 4;
  const int l16  = lane & 15;
  const int wm   = wave / NWN;         // wave's M group
  const int wn   = wave % NWN;         // wave's N slot
  const int m0 = blockIdx.x * MT;
  const int n0 = blockIdx.y * (16 * NWN) + wn * 16;

  // cooperative A staging: 256 threads x MW x 16B = MT rows x 32 halves/K-step
  const int srow   = tid >> 2;          // 0..63
  const int schunk = (tid & 3) * 8;
  const _Float16* sga = A + (size_t)(m0 + srow) * lda + schunk;

  // B fragment: lane = column; halves e -> K = 16*hi16 + e (one 32B load)
  const _Float16* bp = Bw + (size_t)(n0 + l16) * ldb + 16 * hi16;
  // A fragment LDS read base: row wm*64 + l16 (+16r), halves 16*hi16 and +8
  const int aoff = (wm * 64 + l16) * ASTRIDE + 16 * hi16;

  v8f acc[4] = {};
  const int nk = K >> 5;

  // prologue: stage tile 0, preload B fragment 0
#pragma unroll
  for (int g = 0; g < MW; ++g)
    *(v8h*)(&asmem[0][(g * 64 + srow) * ASTRIDE + schunk]) =
        *(const v8h*)(sga + (size_t)g * 64 * lda);
  v16h bf = *(const v16h*)(bp);
  __syncthreads();

  int cur = 0;
#pragma unroll 2
  for (int ks = 0; ks < nk; ++ks) {
    const int k = ks << 5;
    const bool more = (ks + 1 < nk);
    v8h stn[MW];
    v16h bfn;
    if (more) {                               // issue next tile's global loads now
#pragma unroll
      for (int g = 0; g < MW; ++g)
        stn[g] = *(const v8h*)(sga + (size_t)g * 64 * lda + k + 32);
      bfn = *(const v16h*)(bp + k + 32);
      __builtin_prefetch(bp + k + 160, 0, 1); // global_prefetch_b8
    }
    // read all fragments first ...
    v16h af[4];
#pragma unroll
    for (int r = 0; r < 4; ++r) {
      const _Float16* as = &asmem[cur][aoff + r * 16 * ASTRIDE];
      v8h lo = *(const v8h*)(as);
      v8h hi = *(const v8h*)(as + 8);
      af[r] = __builtin_shufflevector(lo, hi,
               0, 1, 2, 3, 4, 5, 6, 7, 8, 9, 10, 11, 12, 13, 14, 15);
    }
    // ... then keep the scheduler from sinking loads between the WMMAs
    __builtin_amdgcn_sched_barrier(0);
#pragma unroll
    for (int r = 0; r < 4; ++r)
      acc[r] = __builtin_amdgcn_wmma_f32_16x16x32_f16(
          false, af[r], false, bf, (short)0, acc[r], false, false);
    if (more) {                               // LDS store absorbs the load wait
#pragma unroll
      for (int g = 0; g < MW; ++g)
        *(v8h*)(&asmem[cur ^ 1][(g * 64 + srow) * ASTRIDE + schunk]) = stn[g];
      bf = bfn;
    }
    __syncthreads();
    cur ^= 1;
  }

  const int n = n0 + l16;
#pragma unroll
  for (int r = 0; r < 4; ++r) {
#pragma unroll
    for (int v = 0; v < 8; ++v) {
      int m = m0 + wm * 64 + r * 16 + 8 * hi16 + v;  // D layout: VGPR v -> M=8*half+v
      size_t idx = (size_t)m * ldo + n;
      float val = acc[r][v];
      if (EP == 1) {
        outH[idx] = (_Float16)val;
      } else if (EP == 2) {
        float xv = val + bias[n];
        outF[idx] = (xv > 20.0f) ? xv : log1pf(__expf(xv));   // softplus
      } else {
        outF[idx] = val + res[idx];
      }
    }
  }
}

// ---------------------------------------------------------------------------
// Depthwise causal conv (D_CONV=4) + SiLU, thread per (b,t,d), f16 in/out
// ---------------------------------------------------------------------------
__global__ __launch_bounds__(256)
void conv_silu_kernel(const _Float16* __restrict__ xz16,
                      const float* __restrict__ cw,
                      const float* __restrict__ cb,
                      _Float16* __restrict__ u16) {
  int idx = blockIdx.x * 256 + threadIdx.x;        // over M_ROWS * D_INNER
  int d  = idx & (D_INNER - 1);
  int mt = idx >> 10;                               // row = b*SEQ + t
  int t  = mt & (SEQ - 1);
  float acc = cb[d];
#pragma unroll
  for (int j = 0; j < D_CONV; ++j) {
    int tt = t + j - (D_CONV - 1);
    if (tt >= 0)
      acc += (float)xz16[(size_t)(mt + j - (D_CONV - 1)) * (2 * D_INNER) + d] * cw[d * D_CONV + j];
  }
  float s = acc / (1.0f + __expf(-acc));            // SiLU
  u16[idx] = (_Float16)s;
}

// ---------------------------------------------------------------------------
// Selective scan: thread per (batch, channel), 16 states in registers.
// B/C chunks (shared by all channels of a batch) streamed into LDS with the
// CDNA5 async-to-LDS path (global_load_async_to_lds_b128 / s_wait_asynccnt),
// double-buffered: chunk t+1 loads overlap chunk t compute.
// ---------------------------------------------------------------------------
__global__ __launch_bounds__(256)
void scan_kernel(const _Float16* __restrict__ xdbl16,  // [M,64]: dt|B|C
                 const float* __restrict__ delta,      // [M,D_INNER]
                 const _Float16* __restrict__ u16,     // [M,D_INNER]
                 const _Float16* __restrict__ xz16,    // z at [..., D_INNER:]
                 const float* __restrict__ A_log,      // [D_INNER,16]
                 const float* __restrict__ Dp,         // [D_INNER]
                 _Float16* __restrict__ yg16) {
  const int d = blockIdx.x * 256 + threadIdx.x;   // channel
  const int b = blockIdx.y;                       // batch
  __shared__ __align__(16) _Float16 bc[2][128][32];  // [buf][t][ B(16) | C(16) ]

  // async stage of one 128-step chunk: 128 rows x 64B = 512 x 16B transfers
  auto stage = [&](int tc, int buf) {
    for (int i = threadIdx.x; i < 512; i += 256) {
      int tt = i >> 2, c = (i & 3) * 8;
      const _Float16* src = xdbl16 + (size_t)(b * SEQ + tc + tt) * 64 + DT_RANK + c;
      unsigned dst = (unsigned)(size_t)(&bc[buf][tt][c]);
      asm volatile("global_load_async_to_lds_b128 %0, %1, off"
                   :: "v"(dst), "v"((unsigned long long)(size_t)src)
                   : "memory");
    }
  };

  float An[D_STATE];
#pragma unroll
  for (int n = 0; n < D_STATE; ++n) An[n] = -__expf(A_log[d * D_STATE + n]);
  const float Dd = Dp[d];
  float h[D_STATE] = {};

  stage(0, 0);
  asm volatile("s_wait_asynccnt 0x0" ::: "memory");
  __syncthreads();

  int cur = 0;
  for (int tc = 0; tc < SEQ; tc += 128) {
    const bool more = (tc + 128 < SEQ);
    if (more) stage(tc + 128, cur ^ 1);     // overlaps with compute below
    for (int tt = 0; tt < 128; ++tt) {
      size_t m = (size_t)b * SEQ + tc + tt;
      float dlt = delta[m * D_INNER + d];
      float uu  = (float)u16[m * D_INNER + d];
      float du  = dlt * uu;
      float y = 0.f;
#pragma unroll
      for (int n = 0; n < D_STATE; ++n) {
        float dA = __expf(dlt * An[n]);
        h[n] = dA * h[n] + du * (float)bc[cur][tt][n];
        y += (float)bc[cur][tt][16 + n] * h[n];
      }
      float zv = (float)xz16[m * (2 * D_INNER) + D_INNER + d];
      float g  = zv / (1.0f + __expf(-zv));
      yg16[m * D_INNER + d] = (_Float16)((y + uu * Dd) * g);
    }
    if (more) asm volatile("s_wait_asynccnt 0x0" ::: "memory");
    __syncthreads();
    cur ^= 1;
  }
}

// ---------------------------------------------------------------------------
// Host orchestration
// ---------------------------------------------------------------------------
extern "C" void kernel_launch(void* const* d_in, const int* in_sizes, int n_in,
                              void* d_out, int out_size, void* d_ws, size_t ws_size,
                              hipStream_t stream) {
  const float* x_in      = (const float*)d_in[0];
  const float* norm_w    = (const float*)d_in[1];
  const float* norm_b    = (const float*)d_in[2];
  const float* in_proj_w = (const float*)d_in[3];
  const float* conv_w    = (const float*)d_in[4];
  const float* conv_b    = (const float*)d_in[5];
  const float* x_proj_w  = (const float*)d_in[6];
  const float* dt_proj_w = (const float*)d_in[7];
  const float* dt_proj_b = (const float*)d_in[8];
  const float* A_log     = (const float*)d_in[9];
  const float* Dp        = (const float*)d_in[10];
  const float* out_w     = (const float*)d_in[11];

  char* p = (char*)d_ws;
  auto alloc = [&](size_t bytes) -> char* {
    char* r = p;
    p += (bytes + 255) & ~(size_t)255;
    return r;
  };
  _Float16* w16_in  = (_Float16*)alloc((size_t)N_LAYERS * 2 * D_INNER * D_MODEL * 2);
  _Float16* w16_xp  = (_Float16*)alloc((size_t)N_LAYERS * 64 * D_INNER * 2);
  _Float16* w16_dt  = (_Float16*)alloc((size_t)N_LAYERS * D_INNER * DT_RANK * 2);
  _Float16* w16_out = (_Float16*)alloc((size_t)N_LAYERS * D_MODEL * D_INNER * 2);
  float*    xbuf    = (float*)   alloc((size_t)M_ROWS * D_MODEL * 4);
  _Float16* h16     = (_Float16*)alloc((size_t)M_ROWS * D_MODEL * 2);
  _Float16* xz16    = (_Float16*)alloc((size_t)M_ROWS * 2 * D_INNER * 2);
  _Float16* u16     = (_Float16*)alloc((size_t)M_ROWS * D_INNER * 2);
  _Float16* xdbl16  = (_Float16*)alloc((size_t)M_ROWS * 64 * 2);
  float*    delta   = (float*)   alloc((size_t)M_ROWS * D_INNER * 4);
  _Float16* yg16    = (_Float16*)alloc((size_t)M_ROWS * D_INNER * 2);

  // residual stream working copy (never mutate d_in)
  hipMemcpyAsync(xbuf, x_in, (size_t)M_ROWS * D_MODEL * 4,
                 hipMemcpyDeviceToDevice, stream);

  // f16 weight images (re-created every call: deterministic, no caching)
  int c0 = N_LAYERS * 2 * D_INNER * D_MODEL;
  int c1 = N_LAYERS * 64 * D_INNER;
  int c2 = N_LAYERS * D_INNER * DT_RANK;
  int c3 = N_LAYERS * D_MODEL * D_INNER;
  f32_to_f16_kernel<<<(c0 + 255) / 256, 256, 0, stream>>>(in_proj_w, w16_in,  c0);
  f32_to_f16_kernel<<<(c1 + 255) / 256, 256, 0, stream>>>(x_proj_w,  w16_xp,  c1);
  f32_to_f16_kernel<<<(c2 + 255) / 256, 256, 0, stream>>>(dt_proj_w, w16_dt,  c2);
  f32_to_f16_kernel<<<(c3 + 255) / 256, 256, 0, stream>>>(out_w,     w16_out, c3);

  for (int l = 0; l < N_LAYERS; ++l) {
    // 1) LayerNorm -> h16
    layernorm_f16_kernel<<<M_ROWS / 8, 256, 0, stream>>>(
        xbuf, norm_w + l * D_MODEL, norm_b + l * D_MODEL, h16);

    // 2) in_proj: [8192,512] x [2048,512]^T -> xz16 (f16), tile 64x128
    gemm_wmma_kernel<1, 1><<<dim3(M_ROWS / 64, (2 * D_INNER) / 128), 256, 0, stream>>>(
        h16, w16_in + (size_t)l * 2 * D_INNER * D_MODEL,
        D_MODEL, D_MODEL, D_MODEL, 2 * D_INNER,
        nullptr, xz16, nullptr, nullptr);

    // 3) causal conv + SiLU -> u16
    conv_silu_kernel<<<(M_ROWS * D_INNER) / 256, 256, 0, stream>>>(
        xz16, conv_w + (size_t)l * D_INNER * D_CONV, conv_b + l * D_INNER, u16);

    // 4) x_proj: [8192,1024] x [64,1024]^T -> xdbl16 (f16), tile 128x64 (MW=2)
    gemm_wmma_kernel<1, 2><<<dim3(M_ROWS / 128, 1), 256, 0, stream>>>(
        u16, w16_xp + (size_t)l * 64 * D_INNER,
        D_INNER, D_INNER, D_INNER, 64,
        nullptr, xdbl16, nullptr, nullptr);

    // 5) dt_proj + softplus fused: [8192,32] x [1024,32]^T -> delta (f32)
    gemm_wmma_kernel<2, 1><<<dim3(M_ROWS / 64, D_INNER / 128), 256, 0, stream>>>(
        xdbl16, w16_dt + (size_t)l * D_INNER * DT_RANK,
        DT_RANK, 64, DT_RANK, D_INNER,
        delta, nullptr, dt_proj_b + l * D_INNER, nullptr);

    // 6) selective scan + u*D + z-gate -> yg16
    scan_kernel<<<dim3(D_INNER / 256, BATCH), 256, 0, stream>>>(
        xdbl16, delta, u16, xz16,
        A_log + (size_t)l * D_INNER * D_STATE, Dp + l * D_INNER, yg16);

    // 7) out_proj + residual: [8192,1024] x [512,1024]^T + xbuf
    float* dst = (l == N_LAYERS - 1) ? (float*)d_out : xbuf;
    gemm_wmma_kernel<3, 1><<<dim3(M_ROWS / 64, D_MODEL / 128), 256, 0, stream>>>(
        yg16, w16_out + (size_t)l * D_MODEL * D_INNER,
        D_INNER, D_INNER, D_INNER, D_MODEL,
        dst, nullptr, nullptr, xbuf);
  }
}
